// ElmoLstm_40089224741432
// MI455X (gfx1250) — compile-verified, matrix operand order checked by reading, not run on hardware
//
#include <hip/hip_runtime.h>

// ---------------------------------------------------------------------------
// ELMo 2-layer bi-LSTM with projection for MI455X (gfx1250, wave32, WMMA).
// All WMMA operands kept bf16-resident in memory: inner K-loops are pure
// global_load_b128 -> v_wmma_f32_16x16x32_bf16 (no per-iteration converts).
// bf16 weights (~150MB) stay resident in the 192MB L2 across all 128 steps.
// ---------------------------------------------------------------------------

typedef __attribute__((ext_vector_type(16))) __bf16 v16bf;
typedef __attribute__((ext_vector_type(8)))  __bf16 v8bf;
typedef __attribute__((ext_vector_type(8)))  float  v8f;

#define B_  32
#define T_  128
#define IN_ 512
#define H_  512
#define C_  4096
#define L_  2
#define G4  (4 * C_)   // 16384 gate rows
#define CLIPV 3.0f

__device__ inline float sigmoidf_(float x) { return 1.0f / (1.0f + __expf(-x)); }

// bf16 fragment (16 elements/lane) per ISA 16-bit A/B layout:
// lanes 0-15: K = {k0..k0+7, k0+16..k0+23}; lanes 16-31: +8 within each half.
__device__ inline v16bf load_frag_bf16(const __bf16* __restrict__ row, int k0, int hi8) {
    v8bf lo = *reinterpret_cast<const v8bf*>(row + k0 + hi8);
    v8bf hi = *reinterpret_cast<const v8bf*>(row + k0 + 16 + hi8);
    return __builtin_shufflevector(lo, hi, 0, 1, 2, 3, 4, 5, 6, 7,
                                   8, 9, 10, 11, 12, 13, 14, 15);
}

// ---------------------------------------------------------------------------
// Kernel 1 per step: gates = x_t@Wx^T + h@Wh^T + bias ; cell update.
// grid = (C/16 column tiles, 2 directions); block = 256 (8 waves).
// Wave w: gate = w&3, mtile = w>>2. Cross-gate combine through LDS.
// ---------------------------------------------------------------------------
__global__ __launch_bounds__(256) void elmo_gates_kernel(
    int layer, int step,
    const __bf16* __restrict__ x_bf,    // [B][T][IN]  layer-0 input (bf16)
    const __bf16* __restrict__ out_bf,  // [L][2][B][T][H] layer outputs (bf16)
    const __bf16* __restrict__ h_bf,    // [L][2][B][H] recurrent state (bf16)
    const __bf16* __restrict__ Wxb,     // [L][2][4C][IN]
    const __bf16* __restrict__ Whb,     // [L][2][4C][H]
    const float* __restrict__ bias,     // [L][2][4C]
    const int*   __restrict__ lengths,  // [B]
    float* __restrict__ c_state,        // [L][2][B][C]
    __bf16* __restrict__ act_bf)        // [2][B][C]  (og*tanh(c_new), bf16)
{
    const int dir   = blockIdx.y;
    const int j0    = blockIdx.x * 16;
    const int t     = dir ? (T_ - 1 - step) : step;
    const int tid   = threadIdx.x;
    const int lane  = tid & 31;
    const int wv    = tid >> 5;
    const int gate  = wv & 3;
    const int mtile = wv >> 2;
    const int hi8   = (lane & 16) ? 8 : 0;
    const int nl    = lane & 15;
    const int ld    = layer * 2 + dir;

    const __bf16* xsrc = (layer == 0)
        ? x_bf
        : (out_bf + (size_t)((layer - 1) * 2 + dir) * B_ * T_ * H_);

    const int bm = mtile * 16 + nl;  // batch row for this lane's A data
    const __bf16* xrow  = xsrc + ((size_t)bm * T_ + t) * IN_;
    const __bf16* hrow  = h_bf + ((size_t)ld * B_ + bm) * H_;
    const int     g     = gate * C_ + j0 + nl;        // gate row (B column)
    const __bf16* wxrow = Wxb + ((size_t)ld * G4 + g) * IN_;
    const __bf16* whrow = Whb + ((size_t)ld * G4 + g) * H_;

    v8f acc = {};
#pragma unroll 4
    for (int kk = 0; kk < IN_ / 32; ++kk) {            // x_t @ Wx^T
        v16bf a = load_frag_bf16(xrow, kk * 32, hi8);
        v16bf b = load_frag_bf16(wxrow, kk * 32, hi8);
        acc = __builtin_amdgcn_wmma_f32_16x16x32_bf16(false, a, false, b,
                                                      (short)0, acc, false, false);
    }
#pragma unroll 4
    for (int kk = 0; kk < H_ / 32; ++kk) {             // h @ Wh^T
        v16bf a = load_frag_bf16(hrow, kk * 32, hi8);
        v16bf b = load_frag_bf16(whrow, kk * 32, hi8);
        acc = __builtin_amdgcn_wmma_f32_16x16x32_bf16(false, a, false, b,
                                                      (short)0, acc, false, false);
    }

    // C/D layout: VGPR r, lanes 0-15 -> M=r, lanes 16-31 -> M=r+8, N=lane&15.
    __shared__ float lds_g[4][32][16];
#pragma unroll
    for (int r = 0; r < 8; ++r) {
        lds_g[gate][mtile * 16 + r + hi8][nl] = acc[r];
    }
    __syncthreads();

    const float* bptr = bias + (size_t)ld * G4;
    for (int e = tid; e < 512; e += 256) {
        const int b = e >> 4;
        const int n = e & 15;
        const int j = j0 + n;
        float ig = lds_g[0][b][n] + bptr[0 * C_ + j];
        float fg = lds_g[1][b][n] + bptr[1 * C_ + j];
        float mi = lds_g[2][b][n] + bptr[2 * C_ + j];
        float og = lds_g[3][b][n] + bptr[3 * C_ + j];
        ig = sigmoidf_(ig); fg = sigmoidf_(fg);
        mi = tanhf(mi);     og = sigmoidf_(og);
        const size_t cidx = ((size_t)ld * B_ + b) * C_ + j;
        const float c_old = c_state[cidx];
        float c_new = fminf(CLIPV, fmaxf(-CLIPV, ig * mi + fg * c_old));
        const bool mask = (t < lengths[b]);
        c_state[cidx] = mask ? c_new : c_old;
        act_bf[((size_t)dir * B_ + b) * C_ + j] = (__bf16)(og * tanhf(c_new));
    }
}

// ---------------------------------------------------------------------------
// Kernel 2 per step: h_new = clip(act @ Wp^T); masked state/output write.
// grid = (H/64, 2 directions); block = 256 (8 waves, each one 16x16 tile).
// Writes fp32 (finalize) and bf16 (next GEMM consumers) copies.
// ---------------------------------------------------------------------------
__global__ __launch_bounds__(256) void elmo_proj_kernel(
    int layer, int step,
    const __bf16* __restrict__ Wpb,     // [L][2][H][C]
    const int*   __restrict__ lengths,
    const __bf16* __restrict__ act_bf,  // [2][B][C]
    float*  __restrict__ h_state,       // [L][2][B][H]  (fp32, for hs output)
    __bf16* __restrict__ h_bf,          // [L][2][B][H]  (bf16, next step A)
    float*  __restrict__ out_buf,       // [L][2][B][T][H] (fp32, finalize)
    __bf16* __restrict__ out_bf)        // [L][2][B][T][H] (bf16, next layer A)
{
    const int dir   = blockIdx.y;
    const int t     = dir ? (T_ - 1 - step) : step;
    const int tid   = threadIdx.x;
    const int lane  = tid & 31;
    const int wv    = tid >> 5;
    const int mtile = wv >> 2;
    const int ntile = blockIdx.x * 4 + (wv & 3);
    const int p0    = ntile * 16;
    const int hi8   = (lane & 16) ? 8 : 0;
    const int nl    = lane & 15;
    const int ld    = layer * 2 + dir;

    const __bf16* arow  = act_bf + ((size_t)dir * B_ + (mtile * 16 + nl)) * C_;
    const __bf16* wprow = Wpb + ((size_t)ld * H_ + p0 + nl) * C_;

    v8f acc = {};
#pragma unroll 4
    for (int kk = 0; kk < C_ / 32; ++kk) {
        v16bf a = load_frag_bf16(arow, kk * 32, hi8);
        v16bf b = load_frag_bf16(wprow, kk * 32, hi8);
        acc = __builtin_amdgcn_wmma_f32_16x16x32_bf16(false, a, false, b,
                                                      (short)0, acc, false, false);
    }

#pragma unroll
    for (int r = 0; r < 8; ++r) {
        const int b = mtile * 16 + r + hi8;
        const int p = p0 + nl;
        const float h_new = fminf(CLIPV, fmaxf(-CLIPV, acc[r]));
        const bool  mask  = (t < lengths[b]);
        const size_t hidx = ((size_t)ld * B_ + b) * H_ + p;
        if (mask) {
            h_state[hidx] = h_new;
            h_bf[hidx]    = (__bf16)h_new;
        }
        const float outv = mask ? h_new : 0.0f;
        const size_t oidx = (((size_t)ld * B_ + b) * T_ + t) * H_ + p;
        out_buf[oidx] = outv;
        out_bf[oidx]  = (__bf16)outv;
    }
}

// ---------------------------------------------------------------------------
// Helpers: fp32->bf16 conversion, zeroing, output assembly.
// ---------------------------------------------------------------------------
__global__ void convert_f32_to_bf16(const float* __restrict__ src,
                                    __bf16* __restrict__ dst, size_t n) {
    size_t i = (size_t)blockIdx.x * blockDim.x + threadIdx.x;
    const size_t stride = (size_t)gridDim.x * blockDim.x;
    for (; i < n; i += stride) dst[i] = (__bf16)src[i];
}

__global__ void zero_f32(float* __restrict__ p, size_t n) {
    size_t i = (size_t)blockIdx.x * blockDim.x + threadIdx.x;
    const size_t stride = (size_t)gridDim.x * blockDim.x;
    for (; i < n; i += stride) p[i] = 0.0f;
}

// d_out = [seq: L,B,T,2H] ++ [hs: L,B,2H] ++ [cs: L,B,2C]; residual on layer 1.
__global__ void elmo_finalize(const float* __restrict__ out_buf,
                              const float* __restrict__ h_state,
                              const float* __restrict__ c_state,
                              float* __restrict__ out, size_t n) {
    const size_t SEQ = (size_t)L_ * B_ * T_ * 2 * H_;
    const size_t HS  = (size_t)L_ * B_ * 2 * H_;
    size_t i = (size_t)blockIdx.x * blockDim.x + threadIdx.x;
    const size_t stride = (size_t)gridDim.x * blockDim.x;
    for (; i < n; i += stride) {
        if (i < SEQ) {
            size_t u = i % (2 * H_);
            size_t rem = i / (2 * H_);
            const int t = (int)(rem % T_); rem /= T_;
            const int b = (int)(rem % B_);
            const int l = (int)(rem / B_);
            const int dir = (u >= H_) ? 1 : 0;
            const int p = (int)u - dir * H_;
            float v = out_buf[(((size_t)(l * 2 + dir) * B_ + b) * T_ + t) * H_ + p];
            if (l != 0)
                v += out_buf[(((size_t)((l - 1) * 2 + dir) * B_ + b) * T_ + t) * H_ + p];
            out[i] = v;
        } else if (i < SEQ + HS) {
            size_t r = i - SEQ;
            const int u = (int)(r % (2 * H_)); r /= (2 * H_);
            const int b = (int)(r % B_);
            const int l = (int)(r / B_);
            const int dir = (u >= H_) ? 1 : 0;
            const int p = u - dir * H_;
            out[i] = h_state[((size_t)(l * 2 + dir) * B_ + b) * H_ + p];
        } else {
            size_t r = i - SEQ - HS;
            const int u = (int)(r % (2 * C_)); r /= (2 * C_);
            const int b = (int)(r % B_);
            const int l = (int)(r / B_);
            const int dir = (u >= C_) ? 1 : 0;
            const int j = u - dir * C_;
            out[i] = c_state[((size_t)(l * 2 + dir) * B_ + b) * C_ + j];
        }
    }
}

// ---------------------------------------------------------------------------
extern "C" void kernel_launch(void* const* d_in, const int* in_sizes, int n_in,
                              void* d_out, int out_size, void* d_ws, size_t ws_size,
                              hipStream_t stream) {
    (void)in_sizes; (void)n_in; (void)ws_size;
    const float* inputs  = (const float*)d_in[0];
    const int*   lengths = (const int*)d_in[1];
    const float* W_x     = (const float*)d_in[2];
    const float* W_h     = (const float*)d_in[3];
    const float* bias    = (const float*)d_in[4];
    const float* W_p     = (const float*)d_in[5];
    float*       out     = (float*)d_out;

    const size_t NWX  = (size_t)L_ * 2 * G4 * IN_;     // 33,554,432
    const size_t NWH  = (size_t)L_ * 2 * G4 * H_;      // 33,554,432
    const size_t NWP  = (size_t)L_ * 2 * H_ * C_;      //  8,388,608
    const size_t NIN  = (size_t)B_ * T_ * IN_;         //  2,097,152
    const size_t NOUT = (size_t)L_ * 2 * B_ * T_ * H_; //  8,388,608
    const size_t NACT = (size_t)2 * B_ * C_;
    const size_t NH   = (size_t)L_ * 2 * B_ * H_;
    const size_t NC   = (size_t)L_ * 2 * B_ * C_;

    char* ws = (char*)d_ws;
    __bf16* Wxb    = (__bf16*)ws; ws += NWX * sizeof(__bf16);
    __bf16* Whb    = (__bf16*)ws; ws += NWH * sizeof(__bf16);
    __bf16* Wpb    = (__bf16*)ws; ws += NWP * sizeof(__bf16);
    __bf16* x_bf   = (__bf16*)ws; ws += NIN * sizeof(__bf16);
    __bf16* out_bf = (__bf16*)ws; ws += NOUT * sizeof(__bf16);
    __bf16* act_bf = (__bf16*)ws; ws += NACT * sizeof(__bf16);
    __bf16* h_bf   = (__bf16*)ws; ws += NH * sizeof(__bf16);
    float* out_buf = (float*)ws;  ws += NOUT * sizeof(float);
    float* h_state = (float*)ws;  ws += NH * sizeof(float);
    float* c_state = (float*)ws;  ws += NC * sizeof(float);   // contiguous after h_state

    convert_f32_to_bf16<<<2048, 256, 0, stream>>>(W_x, Wxb, NWX);
    convert_f32_to_bf16<<<2048, 256, 0, stream>>>(W_h, Whb, NWH);
    convert_f32_to_bf16<<<1024, 256, 0, stream>>>(W_p, Wpb, NWP);
    convert_f32_to_bf16<<<512, 256, 0, stream>>>(inputs, x_bf, NIN);
    zero_f32<<<512, 256, 0, stream>>>(h_state, NH + NC);        // fp32 h and c
    zero_f32<<<64, 256, 0, stream>>>((float*)h_bf, NH / 2);     // bf16 h (as 2x16b)

    for (int l = 0; l < L_; ++l) {
        for (int s = 0; s < T_; ++s) {
            elmo_gates_kernel<<<dim3(C_ / 16, 2), 256, 0, stream>>>(
                l, s, x_bf, out_bf, h_bf, Wxb, Whb, bias, lengths,
                c_state, act_bf);
            elmo_proj_kernel<<<dim3(H_ / 64, 2), 256, 0, stream>>>(
                l, s, Wpb, lengths, act_bf, h_state, h_bf, out_buf, out_bf);
        }
    }

    elmo_finalize<<<2048, 256, 0, stream>>>(out_buf, h_state, c_state, out,
                                            (size_t)out_size);
}
